// ScaledDotProductAttention_40690520162909
// MI455X (gfx1250) — compile-verified, matrix operand order checked by reading, not run on hardware
//
#include <hip/hip_runtime.h>

// Scaled dot-product attention, causal, B=4 S=4096 Dk=64 fp32.
// Outputs: O [B,S,64] then P [B,S,S] concatenated in d_out.

#define BATCH 4
#define SEQ   4096
#define DK    64
#define TQ    16
#define NWAVES 8
#define NTHREADS 256

typedef __attribute__((ext_vector_type(16))) __bf16 v16bf;
typedef __attribute__((ext_vector_type(8)))  float  v8f;
typedef unsigned int tdm_u32x4 __attribute__((ext_vector_type(4)));
typedef int          tdm_i32x8 __attribute__((ext_vector_type(8)));
typedef int          tdm_i32x4 __attribute__((ext_vector_type(4)));

#if defined(__has_builtin)
#if __has_builtin(__builtin_amdgcn_tensor_load_to_lds)
#define HAS_TDM 1
#endif
#if __has_builtin(__builtin_amdgcn_global_store_async_from_lds_b128)
#define HAS_ASYNC_ST 1
#endif
#endif

// Async-store builtin takes (int4 AS1*, int4 AS3*, imm offset, imm cpol).
typedef __attribute__((address_space(1))) tdm_i32x4* g_i4p;
typedef __attribute__((address_space(3))) tdm_i32x4* l_i4p;

#if defined(HAS_TDM)
// 2D TDM load: tile of `rows` x 64 fp32 elements, row stride 64, into LDS.
__device__ __forceinline__ void tdm_load_v_tile(const float* gsrc,
                                                unsigned lds_byte_off,
                                                int rows) {
    const unsigned long long ga = (unsigned long long)(size_t)(const void*)gsrc;
    tdm_u32x4 g0;
    g0[0] = 1u;                                    // count=1, no gather
    g0[1] = lds_byte_off;                          // lds_addr (bytes)
    g0[2] = (unsigned)(ga & 0xffffffffull);        // global_addr[31:0]
    g0[3] = (unsigned)(ga >> 32) | 0x80000000u;    // global_addr[56:32] | type=2
    tdm_i32x8 g1;
    g1[0] = 2 << 16;                               // data_size = 4 bytes
    g1[1] = (DK  & 0xffff) << 16;                  // tensor_dim0 = 64
    g1[2] = (SEQ & 0xffff) << 16;                  // tensor_dim1 = 4096
    g1[3] = DK << 16;                              // tile_dim0 = 64
    g1[4] = rows & 0xffff;                         // tile_dim1 = rows
    g1[5] = DK;                                    // tensor_dim0_stride = 64
    g1[6] = 0;
    g1[7] = 0;
    tdm_i32x4 z4 = {0, 0, 0, 0};
#if __clang_major__ >= 23
    tdm_i32x8 z8 = {0, 0, 0, 0, 0, 0, 0, 0};
    __builtin_amdgcn_tensor_load_to_lds(g0, g1, z4, z4, z8, 0);
#else
    __builtin_amdgcn_tensor_load_to_lds(g0, g1, z4, z4, 0);
#endif
}
#endif

__launch_bounds__(NTHREADS, 1)
__global__ void attn_fwd_kernel(const float* __restrict__ Q,
                                const float* __restrict__ K,
                                const float* __restrict__ V,
                                float* __restrict__ outO,
                                float* __restrict__ outW)
{
    __shared__ float  sc[TQ * SEQ];      // 256 KB score/prob tile (fp32)
    __shared__ __bf16 qs[TQ * DK];       // 2 KB Q tile in bf16
    __shared__ float  red[TQ][16];       // row-reduction partials
    __shared__ float  rowmax[TQ];
    __shared__ float  rowinv[TQ];
    __shared__ float  ored[4][NTHREADS]; // cross-wave O reduction (4 KB)
    __shared__ float  vbuf[2][64 * DK];  // double-buffered V stage (32 KB)

    const int tid  = threadIdx.x;
    const int wave = tid >> 5;
    const int lane = tid & 31;
    const int half = lane >> 4;          // 0: lanes 0-15, 1: lanes 16-31
    const int arow = lane & 15;

    const int blocks_per_b = SEQ / TQ;   // 256
    const int b  = blockIdx.x / blocks_per_b;
    const int qt = blockIdx.x % blocks_per_b;
    const int q_base = qt * TQ;

    const size_t bq = (size_t)b * SEQ + q_base;
    const float* Qb = Q + bq * DK;
    const float* Kb = K + (size_t)b * SEQ * DK;
    const float* Vb = V + (size_t)b * SEQ * DK;

    // ---- Phase 0: load Q tile to LDS as bf16 ----
    for (int i = tid; i < TQ * DK; i += NTHREADS)
        qs[i] = (__bf16)Qb[i];
    __syncthreads();

    // ---- Per-wave A fragments of Q (two K=32 chunks over D=64) ----
    // A layout (16-bit, 16x32): lane<16 holds row=lane, K in {0..7,16..23};
    // lane>=16 holds row=lane-16, K in {8..15,24..31}.
    const int dbase = half * 8;
    v16bf a0, a1;
#pragma unroll
    for (int i = 0; i < 8; ++i) {
        a0[i]     = qs[arow * DK + dbase + i];
        a0[i + 8] = qs[arow * DK + dbase + 16 + i];
        a1[i]     = qs[arow * DK + 32 + dbase + i];
        a1[i + 8] = qs[arow * DK + 32 + dbase + 16 + i];
    }

    // ---- Phase 1: scores = Q K^T / 8 into LDS, skipping fully masked tiles ----
    const int nkt = qt + 1;   // key tiles containing any unmasked column
    for (int kt = wave; kt < nkt; kt += NWAVES) {
        const int k0   = kt * 16;
        const int kcol = k0 + arow;             // B-matrix column = key index
        const float* kp = Kb + (size_t)kcol * DK;
        __builtin_prefetch(kp + (size_t)NWAVES * 16 * DK, 0, 1);
        v16bf b0, b1;
#pragma unroll
        for (int i = 0; i < 8; ++i) {
            b0[i]     = (__bf16)kp[dbase + i];
            b0[i + 8] = (__bf16)kp[dbase + 16 + i];
            b1[i]     = (__bf16)kp[32 + dbase + i];
            b1[i + 8] = (__bf16)kp[32 + dbase + 16 + i];
        }
        v8f c = {};
        c = __builtin_amdgcn_wmma_f32_16x16x32_bf16(false, a0, false, b0,
                                                    (short)0, c, false, false);
        c = __builtin_amdgcn_wmma_f32_16x16x32_bf16(false, a1, false, b1,
                                                    (short)0, c, false, false);
        // C/D layout: vgpr r, lane<16 -> (M=r, N=lane); lane>=16 -> (M=r+8, N=lane-16)
#pragma unroll
        for (int r = 0; r < 8; ++r) {
            const int row = half ? (r + 8) : r;
            sc[row * SEQ + k0 + arow] = c[r] * 0.125f;  // 1/sqrt(64)
        }
    }
    __syncthreads();

    // ---- Phase 2: row max over unmasked columns ----
    const int r  = tid >> 4;       // row 0..15 (16 threads per row)
    const int c0 = tid & 15;
    const int qg = q_base + r;     // global query index; valid keys j <= qg
    {
        float m = -3.0e38f;
        for (int j = c0; j <= qg; j += 16)
            m = fmaxf(m, sc[r * SEQ + j]);
        red[r][c0] = m;
    }
    __syncthreads();
    if (tid < TQ) {
        float mm = red[tid][0];
#pragma unroll
        for (int i = 1; i < 16; ++i) mm = fmaxf(mm, red[tid][i]);
        rowmax[tid] = mm;
    }
    __syncthreads();

    // ---- Phase 3: exp + row sum ----
    {
        const float mrow = rowmax[r];
        float ssum = 0.f;
        for (int j = c0; j <= qg; j += 16) {
            const float e = __expf(sc[r * SEQ + j] - mrow);
            sc[r * SEQ + j] = e;
            ssum += e;
        }
        red[r][c0] = ssum;
    }
    __syncthreads();
    if (tid < TQ) {
        float t = 0.f;
#pragma unroll
        for (int i = 0; i < 16; ++i) t += red[tid][i];
        rowinv[tid] = 1.0f / t;
    }
    __syncthreads();

    // ---- Phase 4a: normalize in LDS (zero masked entries) ----
    {
        const float inv = rowinv[r];
        for (int j = c0 * 4; j < SEQ; j += 64) {
            float4 w;
            w.x = (j + 0 <= qg) ? sc[r * SEQ + j + 0] * inv : 0.f;
            w.y = (j + 1 <= qg) ? sc[r * SEQ + j + 1] * inv : 0.f;
            w.z = (j + 2 <= qg) ? sc[r * SEQ + j + 2] * inv : 0.f;
            w.w = (j + 3 <= qg) ? sc[r * SEQ + j + 3] * inv : 0.f;
            *(float4*)&sc[r * SEQ + j] = w;
        }
    }
    __syncthreads();   // commit DS writes (waits DScnt) before async engine reads LDS

    // ---- Phase 4b: stream P tile LDS -> HBM (the 256 MB roofline stream) ----
    {
        float* wrow = outW + ((size_t)b * SEQ + qg) * SEQ;
#if defined(HAS_ASYNC_ST)
        for (int j = c0 * 4; j < SEQ; j += 64)
            __builtin_amdgcn_global_store_async_from_lds_b128(
                (g_i4p)(wrow + j), (l_i4p)(sc + r * SEQ + j), 0, 0);
#else
        for (int j = c0 * 4; j < SEQ; j += 64)
            *(float4*)&wrow[j] = *(float4*)&sc[r * SEQ + j];
#endif
    }

    // ---- Phase 5: O = P V via WMMA, V staged through LDS by the TDM ----
    const int ntile = wave & 3;              // output d-tile (16 cols each)
    const int khalf = wave >> 2;             // key-half split across wave groups
    const int dcol  = ntile * 16 + arow;     // B column = output feature
    const int kv32  = (q_base + TQ + 31) / 32;   // 32-key chunks with nonzero P
    const int ncp   = (kv32 + 1) >> 1;           // 64-key chunk pairs

#if defined(HAS_TDM)
    if (wave == 0)
        tdm_load_v_tile(Vb, (unsigned)(size_t)(const void*)&vbuf[0][0], 64);
#endif

    v8f oc = {};
    for (int cp = 0; cp < ncp; ++cp) {
        const int buf = cp & 1;
#if defined(HAS_TDM)
        if (wave == 0) {
            if (cp + 1 < ncp) {
                tdm_load_v_tile(Vb + (size_t)(cp + 1) * 64 * DK,
                                (unsigned)(size_t)(const void*)&vbuf[1 - buf][0], 64);
                __builtin_amdgcn_s_wait_tensorcnt(1);  // buffer `buf` ready
            } else {
                __builtin_amdgcn_s_wait_tensorcnt(0);
            }
        }
#else
        for (int i = tid * 4; i < 64 * DK; i += NTHREADS * 4)
            *(float4*)&vbuf[buf][i] =
                *(const float4*)&Vb[(size_t)cp * 64 * DK + i];
#endif
        __syncthreads();   // staged V visible to all waves

        const int ch = cp * 2 + khalf;
        if (ch < kv32) {
            const int kk     = ch * 32;       // global key base of this chunk
            const int klocal = khalf * 32;    // key base inside vbuf
            v16bf pa, vb;
#pragma unroll
            for (int i = 0; i < 8; ++i) {
                pa[i]     = (__bf16)sc[arow * SEQ + kk + dbase + i];
                pa[i + 8] = (__bf16)sc[arow * SEQ + kk + dbase + 16 + i];
            }
#pragma unroll
            for (int i = 0; i < 8; ++i) {
                vb[i]     = (__bf16)vbuf[buf][(klocal + dbase + i) * DK + dcol];
                vb[i + 8] = (__bf16)vbuf[buf][(klocal + dbase + 16 + i) * DK + dcol];
            }
            oc = __builtin_amdgcn_wmma_f32_16x16x32_bf16(false, pa, false, vb,
                                                         (short)0, oc, false, false);
        }
        __syncthreads();   // all waves done with buffer before it is recycled
    }

    // ---- Cross-wave reduction of the two key-halves, then write O ----
    if (khalf == 1) {
#pragma unroll
        for (int rr = 0; rr < 8; ++rr) ored[ntile][rr * 32 + lane] = oc[rr];
    }
    __syncthreads();
    if (khalf == 0) {
#pragma unroll
        for (int rr = 0; rr < 8; ++rr) {
            const float v = oc[rr] + ored[ntile][rr * 32 + lane];
            const int row = half ? (rr + 8) : rr;
            outO[(bq + row) * DK + dcol] = v;
        }
    }
}

extern "C" void kernel_launch(void* const* d_in, const int* in_sizes, int n_in,
                              void* d_out, int out_size, void* d_ws, size_t ws_size,
                              hipStream_t stream) {
    (void)in_sizes; (void)n_in; (void)out_size; (void)d_ws; (void)ws_size;
    const float* Q = (const float*)d_in[0];
    const float* K = (const float*)d_in[1];
    const float* V = (const float*)d_in[2];
    // d_in[3] is the causal mask; causality is applied analytically.
    float* outO = (float*)d_out;                               // [B,S,DK]
    float* outW = (float*)d_out + (size_t)BATCH * SEQ * DK;    // [B,S,S]

    const int nblocks = BATCH * (SEQ / TQ);   // 1024
    attn_fwd_kernel<<<dim3(nblocks), dim3(NTHREADS), 0, stream>>>(Q, K, V, outO, outW);
}
